// MoE_30399778521717
// MI455X (gfx1250) — compile-verified, matrix operand order checked by reading, not run on hardware
//
#include <hip/hip_runtime.h>
#include <hip/hip_bf16.h>
#include <stdint.h>

// ---------------- problem constants (from reference) ----------------
#define NTOK   8192          // B*S = 4*2048
#define DMODEL 1024
#define NEXP   8
#define CAP    NTOK          // a token contributes at most one slot per expert

// ---------------- vector types for WMMA ----------------
typedef __attribute__((ext_vector_type(16))) __bf16 v16bf;
typedef __attribute__((ext_vector_type(8)))  __bf16 v8bf;
typedef __attribute__((ext_vector_type(4)))  __bf16 v4bf;
typedef __attribute__((ext_vector_type(8)))  float  v8f;

#define LDS_AS __attribute__((address_space(3)))

// ---------------- workspace layout (bytes) ----------------
#define OFF_COUNTS 0
#define OFF_LIST   256
#define OFF_WTS    (OFF_LIST + NEXP * CAP * 4)
#define OFF_XBF    (OFF_WTS  + NEXP * CAP * 4)
#define OFF_WBF    (OFF_XBF  + (size_t)NTOK * DMODEL * 2)
// total ~34.1 MB

// ---------------- kernel 1: zero output + counters ----------------
__global__ void moe_zero_init(float4* __restrict__ out4, int n4, int* __restrict__ counts) {
    int i = blockIdx.x * blockDim.x + threadIdx.x;
    if (i < n4) out4[i] = make_float4(0.f, 0.f, 0.f, 0.f);
    if (i < NEXP) counts[i] = 0;
}

// ---------------- kernel 2: fp32 -> bf16 (vectorized x4) ----------------
__global__ void moe_cvt_bf16(const float4* __restrict__ in4, v4bf* __restrict__ out4, int n4) {
    int i = blockIdx.x * blockDim.x + threadIdx.x;
    if (i < n4) {
        float4 f = in4[i];
        v4bf o;
        o[0] = (__bf16)f.x; o[1] = (__bf16)f.y; o[2] = (__bf16)f.z; o[3] = (__bf16)f.w;
        out4[i] = o;
    }
}

// ---------------- kernel 3: gate + softmax + top-2 routing ----------------
__global__ void moe_gate_route(const float* __restrict__ x, const float* __restrict__ Wg,
                               const float* __restrict__ bg, int* __restrict__ counts,
                               int* __restrict__ list, float* __restrict__ wts) {
    const int lane  = threadIdx.x & 31;
    const int wave  = threadIdx.x >> 5;
    const int token = blockIdx.x * 8 + wave;
    if (token >= NTOK) return;

    const float* xr = x + (size_t)token * DMODEL;
    float acc[NEXP];
#pragma unroll
    for (int e = 0; e < NEXP; ++e) acc[e] = 0.f;

    for (int d = lane; d < DMODEL; d += 32) {
        float xv = xr[d];
#pragma unroll
        for (int e = 0; e < NEXP; ++e)
            acc[e] = fmaf(xv, Wg[e * DMODEL + d], acc[e]);
    }
#pragma unroll
    for (int e = 0; e < NEXP; ++e) {
#pragma unroll
        for (int off = 16; off >= 1; off >>= 1)
            acc[e] += __shfl_xor(acc[e], off, 32);
    }

    if (lane == 0) {
        float mx = -1e30f;
#pragma unroll
        for (int e = 0; e < NEXP; ++e) { acc[e] += bg[e]; mx = fmaxf(mx, acc[e]); }
        float p[NEXP], s = 0.f;
#pragma unroll
        for (int e = 0; e < NEXP; ++e) { p[e] = __expf(acc[e] - mx); s += p[e]; }
        float inv = 1.f / s;
#pragma unroll
        for (int e = 0; e < NEXP; ++e) p[e] *= inv;

        int b0 = 0; float v0 = p[0];
#pragma unroll
        for (int e = 1; e < NEXP; ++e) if (p[e] > v0) { v0 = p[e]; b0 = e; }
        int b1 = -1; float v1 = -1.f;
#pragma unroll
        for (int e = 0; e < NEXP; ++e) if (e != b0 && p[e] > v1) { v1 = p[e]; b1 = e; }

        int pos0 = atomicAdd(&counts[b0], 1);
        list[b0 * CAP + pos0] = token;
        wts [b0 * CAP + pos0] = v0;
        int pos1 = atomicAdd(&counts[b1], 1);
        list[b1 * CAP + pos1] = token;
        wts [b1 * CAP + pos1] = v1;
    }
}

// ---------------- CDNA5 async Global->LDS copy (16B per lane), inline asm ----------------
// ISA 15.18.3 opcode 98, GV mode: vdst = per-lane LDS byte offset, vaddr = 64-bit global.
__device__ __forceinline__ void async_copy_b128(const __bf16* gsrc, __bf16* ldst) {
    unsigned lds_off = (unsigned)(uintptr_t)(LDS_AS __bf16*)ldst;   // addrspacecast -> LDS offset
    unsigned long long gaddr = (unsigned long long)(uintptr_t)gsrc;
    asm volatile("global_load_async_to_lds_b128 %0, %1, off"
                 :: "v"(lds_off), "v"(gaddr)
                 : "memory");
}
__device__ __forceinline__ void wait_async_all() {
    asm volatile("s_wait_asynccnt 0x0" ::: "memory");
}

// ---------------- WMMA helpers ----------------
// ISA 7.12.2, 16-bit A/B 16x32: lane L holds row/col (L&15); its 16 bf16 K-values
// are K = kb..kb+7 and kb+16..kb+23 with kb = 8*(L>>4)  -> two 16B contiguous loads.
__device__ __forceinline__ v16bf load_frag_row(const __bf16* p) {
    v8bf lo = *(const v8bf*)(p);
    v8bf hi = *(const v8bf*)(p + 16);
    return __builtin_shufflevector(lo, hi, 0, 1, 2, 3, 4, 5, 6, 7,
                                           8, 9, 10, 11, 12, 13, 14, 15);
}
__device__ __forceinline__ v8f wmma_bf16(v16bf a, v16bf b, v8f c) {
    return __builtin_amdgcn_wmma_f32_16x16x32_bf16(false, a, false, b, (short)0, c, false, false);
}

// ---------------- kernel 4: grouped per-expert GEMM via WMMA ----------------
// grid: (DMODEL/256, CAP/16, NEXP), block: 128 threads = 4 waves.
// A tile (16 tokens x 1024 K, bf16) staged once per block in LDS via async DMA;
// B fragments stream from global, double-buffered (K unrolled by 64).
#define AST (DMODEL + 8)     // LDS row stride in bf16: +16B -> rows hit distinct bank quads

__global__ void __launch_bounds__(128)
moe_gemm_wmma(const __bf16* __restrict__ Xb, const __bf16* __restrict__ Wb,
              const float* __restrict__ be, const int* __restrict__ counts,
              const int* __restrict__ list, const float* __restrict__ wts,
              float* __restrict__ out) {
    const int e   = blockIdx.z;
    const int cnt = counts[e];
    const int tileM = blockIdx.y;
    if (tileM * 16 >= cnt) return;                 // uniform across block

    __shared__ __align__(16) __bf16 Ash[16 * AST];

    const int wave = threadIdx.x >> 5;             // 0..3
    const int lane = threadIdx.x & 31;
    const int colBase = blockIdx.x * 256 + wave * 64;

    // ---- stage A tile into LDS: thread t copies column-chunk t (16B) of all 16 rows ----
    {
        const int cc = threadIdx.x;                // 0..127, 16B chunks along K
#pragma unroll
        for (int r = 0; r < 16; ++r) {
            int slot = tileM * 16 + r;
            if (slot >= cnt) slot = cnt - 1;       // clamp: duplicated rows, results discarded
            const int tok = list[e * CAP + slot];
            async_copy_b128(Xb + (size_t)tok * DMODEL + cc * 8, &Ash[r * AST + cc * 8]);
        }
        wait_async_all();
        __syncthreads();
    }

    // ---- fragment addressing ----
    const int kb    = (lane >> 4) * 8;             // per-lane K chunk base
    const int arow  = lane & 15;
    const __bf16* aLds = &Ash[arow * AST + kb];

    const int ncol0 = colBase + (lane & 15);       // B column = We[e] row (row-major)
    const __bf16* bbase = Wb + (size_t)e * DMODEL * DMODEL + (size_t)ncol0 * DMODEL + kb;

    v8f c0 = {}, c1 = {}, c2 = {}, c3 = {};
    v16bf bA[4], bB[4];
#pragma unroll
    for (int s = 0; s < 4; ++s) bA[s] = load_frag_row(bbase + (size_t)s * 16 * DMODEL);

    for (int k0 = 0; k0 < DMODEL; k0 += 64) {
        // prefetch next half-step while computing with bA
#pragma unroll
        for (int s = 0; s < 4; ++s)
            bB[s] = load_frag_row(bbase + (size_t)s * 16 * DMODEL + (k0 + 32));
        v16bf a0 = load_frag_row(aLds + k0);       // ds_load_b128 x2
        c0 = wmma_bf16(a0, bA[0], c0);
        c1 = wmma_bf16(a0, bA[1], c1);
        c2 = wmma_bf16(a0, bA[2], c2);
        c3 = wmma_bf16(a0, bA[3], c3);

        if (k0 + 64 < DMODEL) {
#pragma unroll
            for (int s = 0; s < 4; ++s)
                bA[s] = load_frag_row(bbase + (size_t)s * 16 * DMODEL + (k0 + 64));
        }
        v16bf a1 = load_frag_row(aLds + k0 + 32);
        c0 = wmma_bf16(a1, bB[0], c0);
        c1 = wmma_bf16(a1, bB[1], c1);
        c2 = wmma_bf16(a1, bB[2], c2);
        c3 = wmma_bf16(a1, bB[3], c3);
    }

    // ---- epilogue: C/D layout (ISA 7.12.2): M = v + 8*(lane>>4), N = lane&15 ----
    const int mhalf = (lane >> 4) * 8;
    const int nl    = lane & 15;
#pragma unroll
    for (int v = 0; v < 8; ++v) {
        const int slot = tileM * 16 + mhalf + v;
        if (slot < cnt) {
            const int   tok = list[e * CAP + slot];
            const float w   = wts [e * CAP + slot];
            float* orow = out + (size_t)tok * DMODEL;
            const int c = colBase + nl;
            atomicAdd(orow + c,      w * (c0[v] + be[e * DMODEL + c]));
            atomicAdd(orow + c + 16, w * (c1[v] + be[e * DMODEL + c + 16]));
            atomicAdd(orow + c + 32, w * (c2[v] + be[e * DMODEL + c + 32]));
            atomicAdd(orow + c + 48, w * (c3[v] + be[e * DMODEL + c + 48]));
        }
    }
}

// ---------------- host launcher ----------------
extern "C" void kernel_launch(void* const* d_in, const int* in_sizes, int n_in,
                              void* d_out, int out_size, void* d_ws, size_t ws_size,
                              hipStream_t stream) {
    (void)in_sizes; (void)n_in; (void)out_size; (void)ws_size;  // top_k (d_in[5]) fixed at 2
    const float* x  = (const float*)d_in[0];   // [B,S,D]
    const float* Wg = (const float*)d_in[1];   // [E,D]
    const float* bg = (const float*)d_in[2];   // [E]
    const float* We = (const float*)d_in[3];   // [E,D,D]
    const float* be = (const float*)d_in[4];   // [E,D]
    float* out = (float*)d_out;

    char* ws = (char*)d_ws;
    int*    counts = (int*)   (ws + OFF_COUNTS);
    int*    list   = (int*)   (ws + OFF_LIST);
    float*  wts    = (float*) (ws + OFF_WTS);
    __bf16* xbf    = (__bf16*)(ws + OFF_XBF);
    __bf16* wbf    = (__bf16*)(ws + OFF_WBF);

    // 1) zero output + per-expert counters
    {
        const int n4 = (NTOK * DMODEL) / 4;
        moe_zero_init<<<(n4 + 255) / 256, 256, 0, stream>>>((float4*)out, n4, counts);
    }
    // 2) fp32 -> bf16 copies of x and We
    {
        const int nx4 = (NTOK * DMODEL) / 4;
        moe_cvt_bf16<<<(nx4 + 255) / 256, 256, 0, stream>>>((const float4*)x,  (v4bf*)xbf, nx4);
        const int nw4 = (NEXP * DMODEL * DMODEL) / 4;
        moe_cvt_bf16<<<(nw4 + 255) / 256, 256, 0, stream>>>((const float4*)We, (v4bf*)wbf, nw4);
    }
    // 3) gate + softmax + top-2 routing (fp32, exact ranking)
    moe_gate_route<<<NTOK / 8, 256, 0, stream>>>(x, Wg, bg, counts, list, wts);

    // 4) grouped per-expert WMMA GEMM (worst-case grid, uniform early exit)
    {
        dim3 grid(DMODEL / 256, CAP / 16, NEXP);   // (4, 512, 8)
        moe_gemm_wmma<<<grid, 128, 0, stream>>>(xbf, wbf, be, counts, list, wts, out);
    }
}
